// EpisodicMemory_64209761075278
// MI455X (gfx1250) — matrix operand order repaired and checked by reading, not compiled, not run on hardware
//
#include <hip/hip_runtime.h>

// ---------------------------------------------------------------------------
// EpisodicMemory fused kernel for MI455X (gfx1250, wave32, WMMA)
//   read_att = softmax(X @ W_att + b_att)        [B, M]
//   read_vec = read_att @ memory                 [B, Dm]
//   cand     = tanh(X @ W_write + b_write)       [B, Dm]
//   new_mem  = memory*(1-u) + u*mean_b(cand),  u = mean_b(read_att)*gate
// B=65536, D_IN=64, M=2048, Dm=64
// ---------------------------------------------------------------------------

typedef _Float16 half_t;
typedef __attribute__((ext_vector_type(16))) _Float16      v16h;
typedef __attribute__((ext_vector_type(8)))  float         v8f;
typedef __attribute__((ext_vector_type(4)))  unsigned int  v4u;

#define BATCH   65536
#define DIN     64
#define MSZ     2048
#define DMEM    64
#define NJ      128     // MSZ / 16 column tiles
#define NKC     64      // MSZ / 32 k-chunks for P@memory

// workspace layout (bytes)
#define WS_WP    0          // packed W_att  B-frags : 256 tiles * 1KB = 256KB
#define WS_MP    262144     // packed memory B-frags : 256 tiles * 1KB = 256KB
#define WS_WWP   524288     // packed W_write B-frags:   8 tiles * 1KB =   8KB
#define WS_ATTP  532480     // att partial colsums   : 1024*2048*4 = 8MB
#define WS_CANDP 8921088    // cand partial colsums  : 1024*64*4  = 256KB
#define WS_WATT  9183232    // reduced write_att     : 2048*4
#define WS_CMEAN 9191424    // reduced cand mean     : 64*4

// dynamic LDS layout (bytes), per block of 4 waves
#define LDS_S(w)   ((w) * 65536)          // 16x2048 f16 logit/prob panel per wave
#define LDS_XH(w)  (262144 + (w) * 2048)  // 16x64 f16 staged X per wave
#define LDS_ATT    270336                 // 2048 f32 block-level att colsum
#define LDS_CAND   278528                 // 64 f32 block-level cand colsum
#define LDS_BYTES  278784

union Frag16 { v16h v; v4u u[2]; };

__device__ inline v16h lds_frag(const char* p0, const char* p1) {
  Frag16 f;
  f.u[0] = *reinterpret_cast<const v4u*>(p0);
  f.u[1] = *reinterpret_cast<const v4u*>(p1);
  return f.v;
}

// ---------------------------------------------------------------------------
// Kernel 0: pack f32 weights into f16 WMMA B-operand fragments.
// B-frag order used here (32x16; A uses the same): lane l, half h -> K=(l/16)*16+h, N=l%16.
// One 1KB block per 32x16 tile; lane's 16 halves are contiguous (32B).
// ---------------------------------------------------------------------------
__global__ __launch_bounds__(32) void k_pack(const float* __restrict__ W_att,
                                             const float* __restrict__ memory,
                                             const float* __restrict__ W_write,
                                             half_t* __restrict__ Wp,
                                             half_t* __restrict__ Mp,
                                             half_t* __restrict__ WWp) {
  const int t = blockIdx.x;
  const int lane = threadIdx.x;
  const int khalf = lane >> 4;   // 0/1 -> K base 0/16 inside tile
  const int col   = lane & 15;   // N within tile
  if (t < 256) {                                   // W_att [64, 2048]
    const int j = t >> 1, kk = t & 1;              // j: col tile, kk: K half (0..31 / 32..63)
    half_t* dst = Wp + (size_t)t * 512 + lane * 16;
    #pragma unroll
    for (int h = 0; h < 16; ++h) {
      const int k = kk * 32 + khalf * 16 + h;
      dst[h] = (half_t)W_att[(size_t)k * MSZ + j * 16 + col];
    }
  } else if (t < 512) {                            // memory [2048, 64]
    const int tt = t - 256;
    const int kc = tt >> 2, n = tt & 3;            // kc: K chunk of 32, n: col tile of Dm
    half_t* dst = Mp + (size_t)tt * 512 + lane * 16;
    #pragma unroll
    for (int h = 0; h < 16; ++h) {
      const int k = kc * 32 + khalf * 16 + h;
      dst[h] = (half_t)memory[(size_t)k * DMEM + n * 16 + col];
    }
  } else {                                         // W_write [64, 64]
    const int tt = t - 512;
    const int kk = tt >> 2, n = tt & 3;
    half_t* dst = WWp + (size_t)tt * 512 + lane * 16;
    #pragma unroll
    for (int h = 0; h < 16; ++h) {
      const int k = kk * 32 + khalf * 16 + h;
      dst[h] = (half_t)W_write[(size_t)k * DMEM + n * 16 + col];
    }
  }
}

// ---------------------------------------------------------------------------
// Kernel 1: fused logits -> softmax -> read_vector (+ partial batch sums).
// 4 waves/block, one 16-row batch tile per wave. Grid = B/64 = 1024 blocks.
// ---------------------------------------------------------------------------
__global__ __launch_bounds__(128) void k_main(const float* __restrict__ x,
                                              const float* __restrict__ b_att,
                                              const float* __restrict__ b_write,
                                              const half_t* __restrict__ Wp,
                                              const half_t* __restrict__ Mp,
                                              const half_t* __restrict__ WWp,
                                              float* __restrict__ out_read,
                                              float* __restrict__ attPart,
                                              float* __restrict__ candPart) {
  extern __shared__ __align__(32) char smem[];
  const int tid  = threadIdx.x;
  const int wave = tid >> 5;
  const int lane = tid & 31;
  const int lrow = lane & 15;   // N column (B/C frags) or M row (A frags)
  const int lhi  = lane >> 4;   // half-wave index

  char*    Sb      = smem + LDS_S(wave);            // 16x2048 f16 panel
  half_t*  Xh      = (half_t*)(smem + LDS_XH(wave));
  float*   att_ls  = (float*)(smem + LDS_ATT);
  float*   cand_ls = (float*)(smem + LDS_CAND);

  for (int i = tid; i < MSZ; i += 128) att_ls[i] = 0.0f;
  if (tid < DMEM) cand_ls[tid] = 0.0f;

  const int rowbase = blockIdx.x * 64 + wave * 16;

  // stage X tile as f16 into LDS (row-major 16x64)
  for (int i = lane; i < 16 * DIN; i += 32)
    Xh[i] = (half_t)x[(size_t)rowbase * DIN + i];
  __syncthreads();

  // A-operand frags of X (16x32 halves each).  A order used here (as packed B): lane l, half h ->
  // M=l%16, K=(l/16)*16+h  => one contiguous 16-half run.
  const char* xb   = (const char*)Xh;
  const int   axof = (lrow * DIN + lhi * 16) * 2;
  const v16h A0 = lds_frag(xb + axof,      xb + axof + 16);
  const v16h A1 = lds_frag(xb + axof + 64, xb + axof + 80);

  // ---- Phase 1: logits = X @ W_att + b_att, track row max, spill f16 panel
  float rmax[8];
  #pragma unroll
  for (int v = 0; v < 8; ++v) rmax[v] = -3.0e38f;

  for (int j = 0; j < NJ; ++j) {
    const v16h B0 = *reinterpret_cast<const v16h*>(Wp + (size_t)(2 * j)     * 512 + lane * 16);
    const v16h B1 = *reinterpret_cast<const v16h*>(Wp + (size_t)(2 * j + 1) * 512 + lane * 16);
    v8f c = {};
    c = __builtin_amdgcn_wmma_f32_16x16x32_f16(false, A0, false, B0, (short)0, c, false, false);
    c = __builtin_amdgcn_wmma_f32_16x16x32_f16(false, A1, false, B1, (short)0, c, false, false);
    const float bb = b_att[j * 16 + lrow];
    #pragma unroll
    for (int v = 0; v < 8; ++v) {        // C layout: row = v + lhi*8, col = j*16+lrow
      const float s = c[v] + bb;
      rmax[v] = fmaxf(rmax[v], s);
      ((half_t*)Sb)[(v + lhi * 8) * MSZ + j * 16 + lrow] = (half_t)s;
    }
  }
  #pragma unroll
  for (int m = 1; m < 16; m <<= 1) {
    #pragma unroll
    for (int v = 0; v < 8; ++v) rmax[v] = fmaxf(rmax[v], __shfl_xor(rmax[v], m, 32));
  }

  // ---- Phase 2: p = exp(s - max), row sums (each lane re-reads its own slots)
  float lsum[8];
  #pragma unroll
  for (int v = 0; v < 8; ++v) lsum[v] = 0.0f;
  for (int j = 0; j < NJ; ++j) {
    #pragma unroll
    for (int v = 0; v < 8; ++v) {
      half_t* p = (half_t*)Sb + (v + lhi * 8) * MSZ + j * 16 + lrow;
      const float e = __expf((float)(*p) - rmax[v]);
      lsum[v] += e;
      *p = (half_t)e;
    }
  }
  #pragma unroll
  for (int m = 1; m < 16; m <<= 1) {
    #pragma unroll
    for (int v = 0; v < 8; ++v) lsum[v] += __shfl_xor(lsum[v], m, 32);
  }
  float invl[8];
  #pragma unroll
  for (int v = 0; v < 8; ++v) invl[v] = 1.0f / lsum[v];

  // ---- Phase 3: column sums of normalized probs -> block partial (write_att)
  for (int j = 0; j < NJ; ++j) {
    float t = 0.0f;
    #pragma unroll
    for (int v = 0; v < 8; ++v)
      t += (float)((half_t*)Sb)[(v + lhi * 8) * MSZ + j * 16 + lrow] * invl[v];
    t += __shfl_xor(t, 16, 32);                       // add other 8 rows
    if (lane < 16) atomicAdd(&att_ls[j * 16 + lane], t);
  }
  __syncthreads();   // panel now read cross-lane below

  // ---- Phase 4: read_vec = P @ memory (A frags straight from LDS panel)
  v8f acc[4] = {v8f{}, v8f{}, v8f{}, v8f{}};
  for (int kc = 0; kc < NKC; ++kc) {
    const int ao = (lrow * MSZ + kc * 32 + lhi * 16) * 2;
    const v16h Ap = lds_frag(Sb + ao, Sb + ao + 16);
    #pragma unroll
    for (int n = 0; n < 4; ++n) {
      const v16h Bm = *reinterpret_cast<const v16h*>(Mp + (size_t)(kc * 4 + n) * 512 + lane * 16);
      acc[n] = __builtin_amdgcn_wmma_f32_16x16x32_f16(false, Ap, false, Bm, (short)0, acc[n], false, false);
    }
  }
  #pragma unroll
  for (int n = 0; n < 4; ++n) {
    #pragma unroll
    for (int v = 0; v < 8; ++v)
      out_read[(size_t)(rowbase + v + lhi * 8) * DMEM + n * 16 + lrow] = acc[n][v] * invl[v];
  }

  // ---- Phase 5: candidate = tanh(X @ W_write + b_write), partial col means
  #pragma unroll
  for (int n = 0; n < 4; ++n) {
    const v16h Bw0 = *reinterpret_cast<const v16h*>(WWp + (size_t)(n)     * 512 + lane * 16);
    const v16h Bw1 = *reinterpret_cast<const v16h*>(WWp + (size_t)(4 + n) * 512 + lane * 16);
    v8f cc = {};
    cc = __builtin_amdgcn_wmma_f32_16x16x32_f16(false, A0, false, Bw0, (short)0, cc, false, false);
    cc = __builtin_amdgcn_wmma_f32_16x16x32_f16(false, A1, false, Bw1, (short)0, cc, false, false);
    const float bw = b_write[n * 16 + lrow];
    float t = 0.0f;
    #pragma unroll
    for (int v = 0; v < 8; ++v) t += tanhf(cc[v] + bw);
    t += __shfl_xor(t, 16, 32);
    if (lane < 16) atomicAdd(&cand_ls[n * 16 + lane], t);
  }
  __syncthreads();

  // ---- write deterministic per-block partials
  for (int i = tid; i < MSZ; i += 128)
    attPart[(size_t)blockIdx.x * MSZ + i] = att_ls[i];
  if (tid < DMEM)
    candPart[(size_t)blockIdx.x * DMEM + tid] = cand_ls[tid];
}

// ---------------------------------------------------------------------------
// Kernel 2a: reduce per-block partials -> write_att [M], cand_mean [Dm]
// ---------------------------------------------------------------------------
__global__ __launch_bounds__(256) void k_reduce(const float* __restrict__ attPart,
                                                const float* __restrict__ candPart,
                                                float* __restrict__ write_att,
                                                float* __restrict__ cand_mean) {
  const int i = blockIdx.x * blockDim.x + threadIdx.x;
  const float invB = 1.0f / (float)BATCH;
  if (i < MSZ) {
    float s = 0.0f;
    for (int wg = 0; wg < 1024; ++wg) s += attPart[(size_t)wg * MSZ + i];
    write_att[i] = s * invB;
  } else if (i < MSZ + DMEM) {
    const int d = i - MSZ;
    float s = 0.0f;
    for (int wg = 0; wg < 1024; ++wg) s += candPart[(size_t)wg * DMEM + d];
    cand_mean[d] = s * invB;
  }
}

// ---------------------------------------------------------------------------
// Kernel 2b: gated memory update
// ---------------------------------------------------------------------------
__global__ __launch_bounds__(256) void k_newmem(const float* __restrict__ memory,
                                                const float* __restrict__ gate,
                                                const float* __restrict__ write_att,
                                                const float* __restrict__ cand_mean,
                                                float* __restrict__ out_mem) {
  const int i = blockIdx.x * blockDim.x + threadIdx.x;
  if (i < MSZ * DMEM) {
    const int m = i >> 6, d = i & 63;
    const float u = write_att[m] * gate[m];
    out_mem[i] = memory[i] * (1.0f - u) + u * cand_mean[d];
  }
}

// ---------------------------------------------------------------------------
extern "C" void kernel_launch(void* const* d_in, const int* in_sizes, int n_in,
                              void* d_out, int out_size, void* d_ws, size_t ws_size,
                              hipStream_t stream) {
  const float* x       = (const float*)d_in[0];
  const float* W_att   = (const float*)d_in[1];
  const float* b_att   = (const float*)d_in[2];
  const float* W_write = (const float*)d_in[3];
  const float* b_write = (const float*)d_in[4];
  const float* memory  = (const float*)d_in[5];
  const float* gate    = (const float*)d_in[6];

  char* ws = (char*)d_ws;
  half_t* Wp        = (half_t*)(ws + WS_WP);
  half_t* Mp        = (half_t*)(ws + WS_MP);
  half_t* WWp       = (half_t*)(ws + WS_WWP);
  float*  attPart   = (float*)(ws + WS_ATTP);
  float*  candPart  = (float*)(ws + WS_CANDP);
  float*  write_att = (float*)(ws + WS_WATT);
  float*  cand_mean = (float*)(ws + WS_CMEAN);

  float* out = (float*)d_out;                 // [B*Dm] read_vector, then [M*Dm] new_memory
  float* out_mem = out + (size_t)BATCH * DMEM;

  k_pack  <<<520, 32, 0, stream>>>(W_att, memory, W_write, Wp, Mp, WWp);
  k_main  <<<BATCH / 64, 128, LDS_BYTES, stream>>>(x, b_att, b_write, Wp, Mp, WWp,
                                                   out, attPart, candPart);
  k_reduce<<<(MSZ + DMEM + 255) / 256, 256, 0, stream>>>(attPart, candPart, write_att, cand_mean);
  k_newmem<<<(MSZ * DMEM + 255) / 256, 256, 0, stream>>>(memory, gate, write_att, cand_mean, out_mem);
}
